// DGLRepresentation_17935783428727
// MI455X (gfx1250) — matrix operand order repaired and checked by reading, not compiled, not run on hardware
//
#include <hip/hip_runtime.h>
#include <hip/hip_bf16.h>

// ---------------------------------------------------------------------------
// CDNA5 WMMA (wave32): D(16x16 f32) = A(16x32 bf16) x B(32x16 bf16) + C
// ---------------------------------------------------------------------------
typedef __attribute__((ext_vector_type(16))) __bf16 v16bf;
typedef __attribute__((ext_vector_type(8)))  float  v8f;

struct FragU { unsigned w[8]; };   // 8 dwords == 16 bf16 == one WMMA operand

__device__ __forceinline__ unsigned short f2bf(float f) {
    unsigned u = __builtin_bit_cast(unsigned, f);
    unsigned r = (u + 0x7FFFu + ((u >> 16) & 1u)) >> 16;   // RNE to bf16
    return (unsigned short)r;
}
__device__ __forceinline__ float bf2f(unsigned short b) {
    return __builtin_bit_cast(float, ((unsigned)b) << 16);
}
__device__ __forceinline__ unsigned pk2bf(float a, float b) {
    return (unsigned)f2bf(a) | ((unsigned)f2bf(b) << 16);
}

// Shared epilogue: C/D layout -> VGPR r, lane-half h: (M = 8h + r, N = nt*16 + lane%16)
__device__ __forceinline__ void gemm_epilogue(const v8f* acc, int mBase, int half, int mLocal,
                                              int M, const float* __restrict__ bias,
                                              float* __restrict__ outF,
                                              unsigned short* __restrict__ outB, int doSilu) {
#pragma unroll
    for (int nt = 0; nt < 8; ++nt) {
#pragma unroll
        for (int r = 0; r < 8; ++r) {
            const int mm = mBase + half * 8 + r;
            const int nn = nt * 16 + mLocal;
            if (mm < M) {
                float v = acc[nt][r] + bias[nn];
                if (doSilu) v = v / (1.0f + __expf(-v));
                const size_t off = (size_t)mm * 128 + nn;
                if (outF) outF[off] = v;
                if (outB) outB[off] = f2bf(v);
            }
        }
    }
}

// ---------------------------------------------------------------------------
// GEMM, A already bf16 [M x 128] (K pre-padded): A fragment = 2x global_load_b128.
// WT is pre-transposed bf16 [n][k] (K-padded); staged to LDS with uint4 copies.
// Block = 256 thr = 8 waves; block tile 128x128; wave tile 16(M) x 128(N).
// ---------------------------------------------------------------------------
__launch_bounds__(256)
__global__ void gemm_a16(const unsigned short* __restrict__ A, int M,
                         const unsigned short* __restrict__ WT,   // [128][128] bf16 [n][k]
                         const float* __restrict__ bias,
                         float* __restrict__ outF, unsigned short* __restrict__ outB,
                         int doSilu) {
    __shared__ __align__(32) unsigned short Wl[128 * 128];   // 32 KB
    const int tid = threadIdx.x, wave = tid >> 5, lane = tid & 31;
    {
        const uint4* s = (const uint4*)WT;
        uint4* d = (uint4*)Wl;
        for (int i = tid; i < 2048; i += 256) d[i] = s[i];
    }
    __syncthreads();

    const int mBase = blockIdx.x * 128 + wave * 16;
    const int half = lane >> 4, mLocal = lane & 15;
    const unsigned short* Arow = A + (size_t)(mBase + mLocal) * 128;
    __builtin_prefetch(Arow, 0, 1);

    v8f acc[8];
#pragma unroll
    for (int t = 0; t < 8; ++t)
#pragma unroll
        for (int j = 0; j < 8; ++j) acc[t][j] = 0.0f;

#pragma unroll
    for (int kt = 0; kt < 4; ++kt) {
        FragU af;
        const int k0 = kt * 32 + 16 * half;
        *(uint4*)&af.w[0] = *(const uint4*)(Arow + k0);        // K run [k0, k0+8)
        *(uint4*)&af.w[4] = *(const uint4*)(Arow + k0 + 8);    // K run [k0+8, k0+16)
        const v16bf aV = __builtin_bit_cast(v16bf, af);
#pragma unroll
        for (int nt = 0; nt < 8; ++nt) {
            const v16bf bV = *(const v16bf*)&Wl[(nt * 16 + mLocal) * 128 + kt * 32 + half * 16];
            acc[nt] = __builtin_amdgcn_wmma_f32_16x16x32_bf16(
                false, aV, false, bV, (short)0, acc[nt], false, false);
        }
    }
    gemm_epilogue(acc, mBase, half, mLocal, M, bias, outF, outB, doSilu);
}

// ---------------------------------------------------------------------------
// GEMM, A fp32 [M x 128] with optional per-row scale (norm_dst fold).
// Branchless float4 A loads; in-place (outF == A) safe (waves own their rows).
// ---------------------------------------------------------------------------
__launch_bounds__(256)
__global__ void gemm_a32(const float* __restrict__ A, int M,
                         const unsigned short* __restrict__ WT,
                         const float* __restrict__ bias,
                         const float* __restrict__ rowScale,   // nullable
                         float* __restrict__ outF, unsigned short* __restrict__ outB,
                         int doSilu) {
    __shared__ __align__(32) unsigned short Wl[128 * 128];
    const int tid = threadIdx.x, wave = tid >> 5, lane = tid & 31;
    {
        const uint4* s = (const uint4*)WT;
        uint4* d = (uint4*)Wl;
        for (int i = tid; i < 2048; i += 256) d[i] = s[i];
    }
    __syncthreads();

    const int mBase = blockIdx.x * 128 + wave * 16;
    const int half = lane >> 4, mLocal = lane & 15;
    const int m = mBase + mLocal;
    const float* Arow = A + (size_t)m * 128;
    __builtin_prefetch(Arow, 0, 1);
    const float scale = rowScale ? rowScale[m] : 1.0f;

    v8f acc[8];
#pragma unroll
    for (int t = 0; t < 8; ++t)
#pragma unroll
        for (int j = 0; j < 8; ++j) acc[t][j] = 0.0f;

#pragma unroll
    for (int kt = 0; kt < 4; ++kt) {
        const int k0 = kt * 32 + 16 * half;
        const float4 r0a = *(const float4*)(Arow + k0);
        const float4 r0b = *(const float4*)(Arow + k0 + 4);
        const float4 r1a = *(const float4*)(Arow + k0 + 8);
        const float4 r1b = *(const float4*)(Arow + k0 + 12);
        FragU af;
        af.w[0] = pk2bf(r0a.x * scale, r0a.y * scale);
        af.w[1] = pk2bf(r0a.z * scale, r0a.w * scale);
        af.w[2] = pk2bf(r0b.x * scale, r0b.y * scale);
        af.w[3] = pk2bf(r0b.z * scale, r0b.w * scale);
        af.w[4] = pk2bf(r1a.x * scale, r1a.y * scale);
        af.w[5] = pk2bf(r1a.z * scale, r1a.w * scale);
        af.w[6] = pk2bf(r1b.x * scale, r1b.y * scale);
        af.w[7] = pk2bf(r1b.z * scale, r1b.w * scale);
        const v16bf aV = __builtin_bit_cast(v16bf, af);
#pragma unroll
        for (int nt = 0; nt < 8; ++nt) {
            const v16bf bV = *(const v16bf*)&Wl[(nt * 16 + mLocal) * 128 + kt * 32 + half * 16];
            acc[nt] = __builtin_amdgcn_wmma_f32_16x16x32_bf16(
                false, aV, false, bV, (short)0, acc[nt], false, false);
        }
    }
    gemm_epilogue(acc, mBase, half, mLocal, M, bias, outF, outB, doSilu);
}

// ---------------------------------------------------------------------------
// One-time prep: weights -> transposed K-padded bf16 [slot][n][k]; x -> padded bf16
// ---------------------------------------------------------------------------
__global__ void prep_weights(const float* __restrict__ w_in, const float* __restrict__ gw,
                             const float* __restrict__ w_out, const float* __restrict__ w_ff,
                             unsigned short* __restrict__ wt) {
    const int idx = blockIdx.x * 256 + threadIdx.x;    // 6 * 16384
    const int s = idx >> 14, r = idx & 16383;
    const int n = r >> 7, k = r & 127;
    const float* W;
    int K = 128;
    switch (s) {
        case 0: W = w_in; K = 74; break;
        case 1: W = gw; break;
        case 2: W = gw + 16384; break;
        case 3: W = gw + 32768; break;
        case 4: W = w_out; break;
        default: W = w_ff; break;
    }
    wt[idx] = (k < K) ? f2bf(W[k * 128 + n]) : (unsigned short)0;
}

__global__ void pad_x(const float* __restrict__ x, unsigned short* __restrict__ xp) {
    const int idx = blockIdx.x * 256 + threadIdx.x;    // N_NODES * 128
    const int node = idx >> 7, k = idx & 127;
    xp[idx] = (k < 74) ? f2bf(x[node * 74 + k]) : (unsigned short)0;
}

// ---------------------------------------------------------------------------
// Zero / degree / norm / scatter / pool
// ---------------------------------------------------------------------------
__global__ void zero_f32v4(float* __restrict__ p, size_t n4) {
    size_t i = (size_t)blockIdx.x * blockDim.x + threadIdx.x;
    const size_t st = (size_t)gridDim.x * blockDim.x;
    const float4 z = make_float4(0.f, 0.f, 0.f, 0.f);
    for (; i < n4; i += st) ((float4*)p)[i] = z;
}

__global__ void count_deg(const int* __restrict__ src, const int* __restrict__ dst,
                          float* __restrict__ degOut, float* __restrict__ degIn, int nE) {
    const int e = blockIdx.x * blockDim.x + threadIdx.x;
    if (e < nE) {
        unsafeAtomicAdd(&degOut[src[e]], 1.0f);
        unsafeAtomicAdd(&degIn[dst[e]], 1.0f);
    }
}

__global__ void finalize_norm(float* __restrict__ a, float* __restrict__ b, int n) {
    const int i = blockIdx.x * blockDim.x + threadIdx.x;
    if (i < n) {
        a[i] = rsqrtf(fmaxf(a[i], 1.0f));
        b[i] = rsqrtf(fmaxf(b[i], 1.0f));
    }
}

// One wave per edge: agg[dst] += norm_src[src] * h_bf16[src]  (128 feats, 4/lane)
__launch_bounds__(256)
__global__ void scatter_edges(const int* __restrict__ src, const int* __restrict__ dst,
                              const float* __restrict__ normSrc,
                              const unsigned short* __restrict__ hB,
                              float* __restrict__ agg, int nE) {
    const int e = blockIdx.x * (blockDim.x >> 5) + (threadIdx.x >> 5);
    const int lane = threadIdx.x & 31;
    const int s = __builtin_amdgcn_readfirstlane(src[e]);
    const int d = __builtin_amdgcn_readfirstlane(dst[e]);
    const float ns = normSrc[s];
    const unsigned long long bits =
        *(const unsigned long long*)(hB + (size_t)s * 128 + lane * 4);
    float* ap = agg + (size_t)d * 128 + lane * 4;
#pragma unroll
    for (int j = 0; j < 4; ++j) {
        const unsigned short b = (unsigned short)(bits >> (16 * j));
        unsafeAtomicAdd(ap + j, bf2f(b) * ns);
    }
}

// One wave per node: pooled[graph_id[n]] += h[n]
__launch_bounds__(256)
__global__ void pool_nodes(const float* __restrict__ h, const int* __restrict__ gid,
                           float* __restrict__ pooled, int nNodes) {
    const int node = blockIdx.x * (blockDim.x >> 5) + (threadIdx.x >> 5);
    const int lane = threadIdx.x & 31;
    const int g = __builtin_amdgcn_readfirstlane(gid[node]);
    const float* hp = h + (size_t)node * 128 + lane * 4;
    float* pp = pooled + (size_t)g * 128 + lane * 4;
#pragma unroll
    for (int j = 0; j < 4; ++j) unsafeAtomicAdd(pp + j, hp[j]);
}

// ---------------------------------------------------------------------------
// Orchestration
// ---------------------------------------------------------------------------
extern "C" void kernel_launch(void* const* d_in, const int* in_sizes, int n_in,
                              void* d_out, int out_size, void* d_ws, size_t ws_size,
                              hipStream_t stream) {
    constexpr int N_NODES = 200000, N_EDGES = 1600000, N_GRAPHS = 10000;
    constexpr int HID = 128, DEPTH = 3;

    const float* x     = (const float*)d_in[0];
    const int*   src   = (const int*)d_in[1];
    const int*   dst   = (const int*)d_in[2];
    const int*   gids  = (const int*)d_in[3];
    const float* w_in  = (const float*)d_in[4];
    const float* b_in  = (const float*)d_in[5];
    const float* gw    = (const float*)d_in[6];   // [3,128,128]
    const float* gb    = (const float*)d_in[7];   // [3,128]
    const float* w_out = (const float*)d_in[8];
    const float* b_out = (const float*)d_in[9];
    const float* w_ff  = (const float*)d_in[10];
    const float* b_ff  = (const float*)d_in[11];
    float* out = (float*)d_out;

    // Workspace layout (256B-aligned offsets)
    char* ws = (char*)d_ws;
    float*          normSrc = (float*)(ws + 0);                   //   0.8 MB
    float*          normDst = (float*)(ws + 800000);              //   0.8 MB
    unsigned short* hB      = (unsigned short*)(ws + 1600000);    //  51.2 MB bf16 acts
    float*          agg     = (float*)(ws + 52800000);            // 102.4 MB fp32 (reused as h_out)
    float*          pooled  = (float*)(ws + 155200000);           //   5.1 MB
    unsigned short* xpad    = (unsigned short*)(ws + 160320000);  //  51.2 MB bf16 padded x
    unsigned short* wtAll   = (unsigned short*)(ws + 211520000);  //   0.2 MB 6x[128][128] bf16

    const size_t nAgg = (size_t)N_NODES * HID;
    const int gemmBlocksN = (N_NODES + 127) / 128;   // 1563
    const int gemmBlocksG = (N_GRAPHS + 127) / 128;  // 79

    // 1) degrees -> norms (norm buffers adjacent: single zero pass)
    zero_f32v4<<<512, 256, 0, stream>>>(normSrc, (size_t)2 * N_NODES / 4);
    count_deg<<<(N_EDGES + 255) / 256, 256, 0, stream>>>(src, dst, normSrc, normDst, N_EDGES);
    finalize_norm<<<(N_NODES + 255) / 256, 256, 0, stream>>>(normSrc, normDst, N_NODES);

    // 2) one-time bf16 prep: weights (transposed, K-padded) + x (K-padded)
    prep_weights<<<6 * 16384 / 256, 256, 0, stream>>>(w_in, gw, w_out, w_ff, wtAll);
    pad_x<<<(int)(nAgg / 256), 256, 0, stream>>>(x, xpad);

    // 3) input embedding: hB = bf16(silu(x @ w_in + b_in))
    gemm_a16<<<gemmBlocksN, 256, 0, stream>>>(
        xpad, N_NODES, wtAll /*slot 0*/, b_in, nullptr, hB, 1);

    // 4) graph conv layers
    for (int l = 0; l < DEPTH; ++l) {
        zero_f32v4<<<4096, 256, 0, stream>>>(agg, nAgg / 4);
        scatter_edges<<<N_EDGES / 8, 256, 0, stream>>>(src, dst, normSrc, hB, agg, N_EDGES);
        const bool last = (l == DEPTH - 1);
        gemm_a32<<<gemmBlocksN, 256, 0, stream>>>(
            agg, N_NODES, wtAll + (size_t)(1 + l) * 16384, gb + (size_t)l * HID,
            normDst, last ? agg : nullptr, last ? nullptr : hB, 1);
    }

    // 5) embedding_out: h = silu(h @ w_out + b_out), in-place fp32
    gemm_a32<<<gemmBlocksN, 256, 0, stream>>>(
        agg, N_NODES, wtAll + (size_t)4 * 16384, b_out, nullptr, agg, nullptr, 1);

    // 6) sum-nodes readout
    zero_f32v4<<<512, 256, 0, stream>>>(pooled, (size_t)N_GRAPHS * HID / 4);
    pool_nodes<<<N_NODES / 8, 256, 0, stream>>>(agg, gids, pooled, N_NODES);

    // 7) final: out = pooled @ w_ff + b_ff (no silu)
    gemm_a32<<<gemmBlocksG, 256, 0, stream>>>(
        pooled, N_GRAPHS, wtAll + (size_t)5 * 16384, b_ff, nullptr, out, nullptr, 0);

    (void)in_sizes; (void)n_in; (void)out_size; (void)ws_size;
}
